// TransformerLayer_24601572672018
// MI455X (gfx1250) — matrix operand order repaired and checked by reading, not compiled, not run on hardware
//
#include <hip/hip_runtime.h>
#include <hip/hip_bf16.h>

typedef __attribute__((ext_vector_type(16))) _Float16 v16h;
typedef __attribute__((ext_vector_type(8)))  _Float16 v8h;
typedef __attribute__((ext_vector_type(8)))  float    v8f;
typedef __attribute__((ext_vector_type(4)))  unsigned v4u;
typedef __attribute__((ext_vector_type(8)))  int      v8i;
typedef __attribute__((ext_vector_type(4)))  int      v4i;

#define D_MODEL 1024
#define D_INT   256
#define NBATCH  4
#define SEQ     2048
#define MTOT    (NBATCH * SEQ)

// GEMM tiling: block = 256 thr = 8 waves; block tile 64(M) x 128(N); K-step 32.
#define A_ROWS   64
#define B_ROWS   128
#define TILE_ROWS (A_ROWS + B_ROWS)
// LDS row stride: 32 halfs (64B) + TDM pad of 4 dwords (16B) -> 80B = 40 halfs.
// Keeps ds_load_b128 16B-aligned and spreads rows across banks (20*r mod 64).
#define ROWPAD   40

#if __has_builtin(__builtin_amdgcn_tensor_load_to_lds)
#define USE_TDM 1
#else
#define USE_TDM 0
#endif

// ---------------------------------------------------------------------------
// WMMA helper (gfx1250: V_WMMA_F32_16X16X32_F16, wave32)
// ---------------------------------------------------------------------------
__device__ __forceinline__ v8f wmma_f16(v16h a, v16h b, v8f c) {
  return __builtin_amdgcn_wmma_f32_16x16x32_f16(false, a, false, b, (short)0, c,
                                                false, false);
}

// A fragment from LDS tile (rows = M, 40-half row stride).
// Lane L: row = rowbase + (L&15), h = L>>4; elems 0..7 -> K=h*8+0..7,
// 8..15 -> K=16+h*8+0..7 (16-bit A layout, ISA 7.12.2).
__device__ __forceinline__ v16h lds_a_frag(const _Float16* base, int rowbase,
                                           int lane) {
  const _Float16* p = base + (rowbase + (lane & 15)) * ROWPAD + (lane >> 4) * 8;
  v8h lo = *(const v8h*)(p);
  v8h hi = *(const v8h*)(p + 16);
  v16h r;
#pragma unroll
  for (int i = 0; i < 8; ++i) { r[i] = lo[i]; r[i + 8] = hi[i]; }
  return r;
}

// B fragment from LDS tile (rows = N of Bt, i.e. B column-major source).
// Lane L: col n = rowbase + (L&15), h = L>>4; elems 0..7 -> K=h*8+0..7,
// 8..15 -> K=16+h*8+0..7 (same as A).
__device__ __forceinline__ v16h lds_b_frag(const _Float16* base, int rowbase,
                                           int lane) {
  const _Float16* p = base + (rowbase + (lane & 15)) * ROWPAD + (lane >> 4) * 8;
  v8h lo = *(const v8h*)(p);
  v8h hi = *(const v8h*)(p + 16);
  v16h r;
#pragma unroll
  for (int i = 0; i < 8; ++i) { r[i] = lo[i]; r[i + 8] = hi[i]; }
  return r;
}

#if USE_TDM
// Issue one TDM 2-D tile load: rows x 32 halfs from global (row stride ld
// halfs) into LDS at ldsByteAddr, packed rows with 16B pad per row.
// D# packing per CDNA5 ISA §8.3/§8.4.
__device__ __forceinline__ void tdm_issue(const _Float16* gsrc, int ld,
                                          unsigned ldsByteAddr, unsigned rows) {
  unsigned long long ga = (unsigned long long)(size_t)gsrc;
  v4u g0;
  g0[0] = 1u;                                    // count=1 (valid), user mode
  g0[1] = ldsByteAddr;                           // lds_addr
  g0[2] = (unsigned)(ga & 0xffffffffu);          // global_addr[31:0]
  g0[3] = (unsigned)((ga >> 32) & 0x01ffffffu)   // global_addr[56:32]
        | (2u << 30);                            // type=2 ("image")
  const unsigned td0 = 32u, tile0 = 32u;
  unsigned long long s0 = (unsigned long long)(unsigned)ld;  // dim0 stride
  v8i g1;
  g1[0] = (int)((1u << 16)      // data_size = 2 bytes
              | (1u << 20)      // pad_enable
              | (3u << 22)      // pad_interval: every 16 dwords (one row)
              | (3u << 25));    // pad_amount: 4 dwords (16B)
  g1[1] = (int)((td0 & 0xffffu) << 16);                 // tensor_dim0[15:0]
  g1[2] = (int)((td0 >> 16) | ((rows & 0xffffu) << 16));// td0 hi | tensor_dim1 lo
  g1[3] = (int)((rows >> 16) | (tile0 << 16));          // td1 hi | tile_dim0
  g1[4] = (int)(rows & 0xffffu);                        // tile_dim1 (tile_dim2=0)
  g1[5] = (int)(s0 & 0xffffffffu);                      // dim0_stride[31:0]
  g1[6] = (int)((s0 >> 32) & 0xffffu);                  // stride hi | dim1_stride=0
  g1[7] = 0;
  v4i gz = {0, 0, 0, 0};
#if defined(__clang_major__) && __clang_major__ >= 23
  v8i gz8 = {0, 0, 0, 0, 0, 0, 0, 0};
  __builtin_amdgcn_tensor_load_to_lds(g0, g1, gz, gz, gz8, 0);
#else
  __builtin_amdgcn_tensor_load_to_lds(g0, g1, gz, gz, 0);
#endif
}
#endif

// ---------------------------------------------------------------------------
// Generic batched WMMA GEMM, LDS-staged, TDM double-buffered:
//   C[z][m][n] = sum_k A[z][m][k] * Bt[z][n][k]  (+bias[n]) (relu) (+resid)
// grid = (N/128, M/64, Z)
// ---------------------------------------------------------------------------
__global__ __launch_bounds__(256)
void gemm_wmma_kernel(const _Float16* __restrict__ A,
                      const _Float16* __restrict__ Bt,
                      const float* __restrict__ bias,
                      const float* __restrict__ resid,
                      float* __restrict__ outF,
                      _Float16* __restrict__ outH,
                      _Float16* __restrict__ outHT,
                      int K, int lda, int ldb, int ldc, long ldt,
                      long batchA, long batchB, long batchC, long batchR,
                      int relu) {
  __shared__ alignas(128) _Float16 smem[2][TILE_ROWS * ROWPAD];

  const int z = blockIdx.z;
  const _Float16* Ab = A + (size_t)z * batchA + (size_t)(blockIdx.y * 64) * lda;
  const _Float16* Bb = Bt + (size_t)z * batchB + (size_t)(blockIdx.x * 128) * ldb;

  const int lane = threadIdx.x & 31;
  const int wave = threadIdx.x >> 5;
  const int wm = wave & 1;       // 2 waves along M
  const int wn = wave >> 1;      // 4 waves along N
  const int T = K / 32;

  v8f acc00 = {}, acc01 = {}, acc10 = {}, acc11 = {};

#if USE_TDM
  // Wave 0 drives the Tensor Data Mover; TENSORcnt tracks the DMAs.
  if (wave == 0) {
    unsigned ldsA0 = (unsigned)(size_t)&smem[0][0];
    tdm_issue(Ab, lda, ldsA0, A_ROWS);
    tdm_issue(Bb, ldb, ldsA0 + A_ROWS * ROWPAD * 2, B_ROWS);
  }
#else
  {  // cooperative fallback: 768 16B chunks, 3 per thread
    const int t = threadIdx.x;
#pragma unroll
    for (int c = t; c < TILE_ROWS * 4; c += 256) {
      int row = c >> 2, part = c & 3;
      const _Float16* src = (row < A_ROWS)
          ? (Ab + (size_t)row * lda + part * 8)
          : (Bb + (size_t)(row - A_ROWS) * ldb + part * 8);
      *(v8h*)(&smem[0][row * ROWPAD + part * 8]) = *(const v8h*)src;
    }
  }
#endif

  for (int i = 0; i < T; ++i) {
    const int buf = i & 1;
    const bool hasNext = (i + 1) < T;
#if USE_TDM
    if (wave == 0) {
      if (hasNext) {
        unsigned ldsN = (unsigned)(size_t)&smem[buf ^ 1][0];
        tdm_issue(Ab + (i + 1) * 32, lda, ldsN, A_ROWS);
        tdm_issue(Bb + (i + 1) * 32, ldb, ldsN + A_ROWS * ROWPAD * 2, B_ROWS);
        __builtin_amdgcn_s_wait_tensorcnt((short)2);  // tile i landed
      } else {
        __builtin_amdgcn_s_wait_tensorcnt((short)0);
      }
    }
#else
    if (hasNext) {
      const int t = threadIdx.x;
      const int k0 = (i + 1) * 32;
#pragma unroll
      for (int c = t; c < TILE_ROWS * 4; c += 256) {
        int row = c >> 2, part = c & 3;
        const _Float16* src = (row < A_ROWS)
            ? (Ab + (size_t)row * lda + k0 + part * 8)
            : (Bb + (size_t)(row - A_ROWS) * ldb + k0 + part * 8);
        *(v8h*)(&smem[buf ^ 1][row * ROWPAD + part * 8]) = *(const v8h*)src;
      }
    }
#endif
    __syncthreads();   // tile i visible to all waves

    const _Float16* sA = &smem[buf][0];
    const _Float16* sB = &smem[buf][A_ROWS * ROWPAD];
    v16h a0 = lds_a_frag(sA, wm * 32,      lane);
    v16h a1 = lds_a_frag(sA, wm * 32 + 16, lane);
    v16h b0 = lds_b_frag(sB, wn * 32,      lane);
    v16h b1 = lds_b_frag(sB, wn * 32 + 16, lane);
    acc00 = wmma_f16(a0, b0, acc00);
    acc01 = wmma_f16(a0, b1, acc01);
    acc10 = wmma_f16(a1, b0, acc10);
    acc11 = wmma_f16(a1, b1, acc11);

    __syncthreads();   // all waves done with buf before it is overwritten
  }

  // Epilogue. C/D layout: VGPR r -> (m = r + 8*(lane>>4), n = lane&15).
  const int mbase = blockIdx.y * 64 + wm * 32;
  const int nbase = blockIdx.x * 128 + wn * 32;
  const int h  = lane >> 4;
  const int nl = lane & 15;
#pragma unroll
  for (int tm = 0; tm < 2; ++tm) {
#pragma unroll
    for (int tn = 0; tn < 2; ++tn) {
      const v8f* accp = (tm == 0) ? ((tn == 0) ? &acc00 : &acc01)
                                  : ((tn == 0) ? &acc10 : &acc11);
      v8f acc = *accp;
      const int nn = nbase + tn * 16 + nl;
      const float bv = bias ? bias[nn] : 0.0f;
#pragma unroll
      for (int r = 0; r < 8; ++r) {
        const int mm = mbase + tm * 16 + r + 8 * h;
        float val = acc[r] + bv;
        if (relu) val = fmaxf(val, 0.0f);
        if (resid) val += resid[(size_t)z * batchR + (size_t)mm * ldc + nn];
        const size_t co = (size_t)z * batchC + (size_t)mm * ldc + nn;
        if (outF)  outF[co] = val;
        if (outH)  outH[co] = (_Float16)val;
        if (outHT) outHT[(size_t)nn * ldt + mm] = (_Float16)val;  // unbatched use
      }
    }
  }
}

// ---------------------------------------------------------------------------
// Row softmax with causal mask + 1/sqrt(Di) scale. One block (256 thr) / row.
// ---------------------------------------------------------------------------
__global__ __launch_bounds__(256)
void softmax_kernel(float* __restrict__ attnF, _Float16* __restrict__ attnH) {
  const int row = blockIdx.x;
  const int q   = row % SEQ;
  float* rowF = attnF + (size_t)row * SEQ;
  _Float16* rowH = attnH + (size_t)row * SEQ;
  const int t = threadIdx.x;
  const float scale = 0.0625f;  // 1/sqrt(256)

  __shared__ float red[256];

  float mx = -INFINITY;
  for (int j = t; j < SEQ; j += 256) {
    float v = (j <= q) ? rowF[j] * scale : -INFINITY;
    mx = fmaxf(mx, v);
  }
  red[t] = mx;
  __syncthreads();
  for (int s = 128; s > 0; s >>= 1) {
    if (t < s) red[t] = fmaxf(red[t], red[t + s]);
    __syncthreads();
  }
  mx = red[0];
  __syncthreads();

  float sum = 0.0f;
  for (int j = t; j < SEQ; j += 256) {
    float v = (j <= q) ? expf(rowF[j] * scale - mx) : 0.0f;
    rowF[j] = v;
    sum += v;
  }
  red[t] = sum;
  __syncthreads();
  for (int s = 128; s > 0; s >>= 1) {
    if (t < s) red[t] += red[t + s];
    __syncthreads();
  }
  const float inv = 1.0f / red[0];

  for (int j = t; j < SEQ; j += 256) {
    float v = rowF[j] * inv;
    rowF[j] = v;
    rowH[j] = (_Float16)v;
  }
}

// ---------------------------------------------------------------------------
// Elementwise converts
// ---------------------------------------------------------------------------
__global__ __launch_bounds__(256)
void f32_to_f16_kernel(const float* __restrict__ in, _Float16* __restrict__ out,
                       long n) {
  long i = (long)blockIdx.x * 256 + threadIdx.x;
  if (i < n) out[i] = (_Float16)in[i];
}

// W [K,N] f32 row-major -> Wt [N,K] f16 row-major
__global__ __launch_bounds__(256)
void transpose_to_f16_kernel(const float* __restrict__ W,
                             _Float16* __restrict__ Wt, int K, int N) {
  long i = (long)blockIdx.x * 256 + threadIdx.x;
  if (i < (long)K * N) {
    int k = (int)(i / N);
    int n = (int)(i % N);
    Wt[(size_t)n * K + k] = (_Float16)W[i];
  }
}

// ---------------------------------------------------------------------------
// Orchestration
// ---------------------------------------------------------------------------
extern "C" void kernel_launch(void* const* d_in, const int* in_sizes, int n_in,
                              void* d_out, int out_size, void* d_ws, size_t ws_size,
                              hipStream_t stream) {
  (void)in_sizes; (void)n_in; (void)out_size; (void)ws_size;
  const float* x  = (const float*)d_in[0];
  const float* Wq = (const float*)d_in[1];
  const float* bq = (const float*)d_in[2];
  const float* Wk = (const float*)d_in[3];
  const float* bk = (const float*)d_in[4];
  const float* Wv = (const float*)d_in[5];
  const float* bv = (const float*)d_in[6];
  const float* W1 = (const float*)d_in[7];
  const float* b1 = (const float*)d_in[8];
  const float* W2 = (const float*)d_in[9];
  const float* b2 = (const float*)d_in[10];

  float* outF  = (float*)d_out;                   // [B,S,D] h+ff
  float* attnF = outF + (size_t)MTOT * D_MODEL;   // [B,S,S] attn

  char* ws = (char*)d_ws;
  size_t off = 0;
  auto carve = [&](size_t bytes) -> char* {
    char* p = ws + off;
    off = (off + bytes + 255) & ~(size_t)255;
    return p;
  };
  _Float16* xh    = (_Float16*)carve((size_t)MTOT * D_MODEL * 2);  // also h_h
  _Float16* Wqt   = (_Float16*)carve((size_t)D_MODEL * D_INT * 2);
  _Float16* Wkt   = (_Float16*)carve((size_t)D_MODEL * D_INT * 2);
  _Float16* Wvt   = (_Float16*)carve((size_t)D_MODEL * D_MODEL * 2);
  _Float16* W1t   = (_Float16*)carve((size_t)D_MODEL * D_MODEL * 2);
  _Float16* W2t   = (_Float16*)carve((size_t)D_MODEL * D_MODEL * 2);
  _Float16* qh    = (_Float16*)carve((size_t)MTOT * D_INT * 2);
  _Float16* kh    = (_Float16*)carve((size_t)MTOT * D_INT * 2);
  _Float16* vT    = (_Float16*)carve((size_t)D_MODEL * MTOT * 2);  // also relu_h
  _Float16* attnH = (_Float16*)carve((size_t)NBATCH * SEQ * SEQ * 2);
  float*    hF    = (float*)   carve((size_t)MTOT * D_MODEL * 4);
  _Float16* hh    = xh;  // reuse: x f16 dead after v GEMM
  _Float16* rh    = vT;  // reuse: vT dead after attn@v GEMM

  const long S2  = (long)SEQ * SEQ;
  const long SD  = (long)SEQ * D_MODEL;
  const long SDi = (long)SEQ * D_INT;

  // 0) converts
  f32_to_f16_kernel<<<(MTOT * D_MODEL) / 256, 256, 0, stream>>>(x, xh, (long)MTOT * D_MODEL);
  transpose_to_f16_kernel<<<(D_MODEL * D_INT) / 256, 256, 0, stream>>>(Wq, Wqt, D_MODEL, D_INT);
  transpose_to_f16_kernel<<<(D_MODEL * D_INT) / 256, 256, 0, stream>>>(Wk, Wkt, D_MODEL, D_INT);
  transpose_to_f16_kernel<<<(D_MODEL * D_MODEL) / 256, 256, 0, stream>>>(Wv, Wvt, D_MODEL, D_MODEL);
  transpose_to_f16_kernel<<<(D_MODEL * D_MODEL) / 256, 256, 0, stream>>>(W1, W1t, D_MODEL, D_MODEL);
  transpose_to_f16_kernel<<<(D_MODEL * D_MODEL) / 256, 256, 0, stream>>>(W2, W2t, D_MODEL, D_MODEL);

  // 1) q = x@Wq + bq -> f16 [MTOT, 256]
  gemm_wmma_kernel<<<dim3(D_INT / 128, MTOT / 64, 1), 256, 0, stream>>>(
      xh, Wqt, bq, nullptr, nullptr, qh, nullptr,
      D_MODEL, D_MODEL, D_MODEL, D_INT, 0, 0, 0, 0, 0, 0);
  // 2) k = x@Wk + bk -> f16 [MTOT, 256]
  gemm_wmma_kernel<<<dim3(D_INT / 128, MTOT / 64, 1), 256, 0, stream>>>(
      xh, Wkt, bk, nullptr, nullptr, kh, nullptr,
      D_MODEL, D_MODEL, D_MODEL, D_INT, 0, 0, 0, 0, 0, 0);
  // 3) v = x@Wv + bv -> f16 transposed vT [D, MTOT]
  gemm_wmma_kernel<<<dim3(D_MODEL / 128, MTOT / 64, 1), 256, 0, stream>>>(
      xh, Wvt, bv, nullptr, nullptr, nullptr, vT,
      D_MODEL, D_MODEL, D_MODEL, D_MODEL, (long)MTOT, 0, 0, 0, 0, 0);
  // 4) scores[b] = q[b] @ k[b]^T -> f32 into d_out attn region (unscaled)
  gemm_wmma_kernel<<<dim3(SEQ / 128, SEQ / 64, NBATCH), 256, 0, stream>>>(
      qh, kh, nullptr, nullptr, attnF, nullptr, nullptr,
      D_INT, D_INT, D_INT, SEQ, 0, SDi, SDi, S2, 0, 0);
  // 5) softmax (scale + causal mask), in place f32 + f16 copy
  softmax_kernel<<<MTOT, 256, 0, stream>>>(attnF, attnH);
  // 6) h[b] = x[b] + attn[b] @ v[b] -> f32 hF and f16 hh
  gemm_wmma_kernel<<<dim3(D_MODEL / 128, SEQ / 64, NBATCH), 256, 0, stream>>>(
      attnH, vT, nullptr, x, hF, hh, nullptr,
      SEQ, SEQ, MTOT, D_MODEL, 0, S2, (long)SEQ, SD, SD, 0);
  // 7) r = relu(h@W1 + b1) -> f16
  gemm_wmma_kernel<<<dim3(D_MODEL / 128, MTOT / 64, 1), 256, 0, stream>>>(
      hh, W1t, b1, nullptr, nullptr, rh, nullptr,
      D_MODEL, D_MODEL, D_MODEL, D_MODEL, 0, 0, 0, 0, 0, 1);
  // 8) out = h + (r@W2 + b2) -> f32 d_out
  gemm_wmma_kernel<<<dim3(D_MODEL / 128, MTOT / 64, 1), 256, 0, stream>>>(
      rh, W2t, b2, hF, outF, nullptr, nullptr,
      D_MODEL, D_MODEL, D_MODEL, D_MODEL, 0, 0, 0, 0, SD, 0);
}
